// MultiChevB_26645977104436
// MI455X (gfx1250) — compile-verified, matrix operand order checked by reading, not run on hardware
//
#include <hip/hip_runtime.h>
#include <hip/hip_bf16.h>

typedef __attribute__((ext_vector_type(16))) _Float16 v16h;
typedef __attribute__((ext_vector_type(8)))  float    v8f;

#define NN      100000      // nodes
#define IC      64          // in channels
#define OC      100         // out channels per head
#define NE      1600000     // edges
#define NCOL    300         // total output cols
#define NT_N    19          // 304/16 col tiles
#define NT_M    6250        // 100000/16 row tiles
#define NKT     8           // 256/32 K steps
#define TOTAL_TILES (NT_M * NT_N)

// ---------------- edge normalization ----------------

__global__ void deg_kernel(const long long* __restrict__ ei,
                           const float* __restrict__ ew,
                           float* __restrict__ deg) {
    int e = blockIdx.x * blockDim.x + threadIdx.x;
    if (e >= NE) return;
    long long r = ei[e];
    long long c = ei[NE + e];
    if (r != c) atomicAdd(&deg[r], ew[e]);
}

__global__ void norm_kernel(const long long* __restrict__ ei,
                            const float* __restrict__ ew,
                            const float* __restrict__ deg,
                            float* __restrict__ nrm) {
    int e = blockIdx.x * blockDim.x + threadIdx.x;
    if (e >= NE) return;
    long long r = ei[e];
    long long c = ei[NE + e];
    float w  = (r != c) ? ew[e] : 0.0f;
    float dr = deg[r];
    float dc = deg[c];
    float ir = (dr > 0.0f) ? rsqrtf(dr) : 0.0f;
    float ic = (dc > 0.0f) ? rsqrtf(dc) : 0.0f;
    nrm[e] = -ir * w * ic;
}

// ---------------- Chebyshev propagation ----------------

// dst = -src (elementwise, float4-vectorized); used to seed T_k = -T_{k-2}
__global__ void negcopy_kernel(const float* __restrict__ src,
                               float* __restrict__ dst, int n4) {
    int i = blockIdx.x * blockDim.x + threadIdx.x;
    if (i >= n4) return;
    float4 v = reinterpret_cast<const float4*>(src)[i];
    reinterpret_cast<float4*>(dst)[i] = make_float4(-v.x, -v.y, -v.z, -v.w);
}

// dst[col] += scale * norm * h[row]; 16 threads per edge, float4 per thread
__global__ void scatter_kernel(const long long* __restrict__ ei,
                               const float* __restrict__ nrm,
                               const float* __restrict__ h,
                               float* __restrict__ dst, float scale) {
    long long tid = (long long)blockIdx.x * blockDim.x + threadIdx.x;
    if (tid >= (long long)NE * 16) return;
    int e = (int)(tid >> 4);
    int g = (int)(tid & 15);
    float w = nrm[e] * scale;
    if (w == 0.0f) return;       // self-loops & zero-degree edges contribute 0
    int r = (int)ei[e];
    int c = (int)ei[NE + e];
    float4 hv = *reinterpret_cast<const float4*>(h + (size_t)r * IC + g * 4);
    float* d = dst + (size_t)c * IC + g * 4;
    atomicAdd(d + 0, w * hv.x);
    atomicAdd(d + 1, w * hv.y);
    atomicAdd(d + 2, w * hv.z);
    atomicAdd(d + 3, w * hv.w);
}

// ---------------- weight packing into B-fragment order ----------------

// logical Wbig[k (0..255)][n (0..303)]; zero blocks where head order < k-index
__device__ __forceinline__ float wbig_val(int k, int n,
                                          const float* W1, const float* W2,
                                          const float* W3) {
    int t  = k >> 6;      // which Chebyshev order
    int kl = k & 63;      // input channel
    if (n < 100)      return (t < 2) ? W1[(t * IC + kl) * OC + n]        : 0.0f;
    else if (n < 200) return (t < 3) ? W2[(t * IC + kl) * OC + (n - 100)]: 0.0f;
    else if (n < 300) return           W3[(t * IC + kl) * OC + (n - 200)];
    return 0.0f;          // pad cols 300..303
}

// Pack into exact v_wmma 16-bit B layout: contiguous 16-half blob per lane.
__global__ void pack_b_kernel(const float* __restrict__ W1,
                              const float* __restrict__ W2,
                              const float* __restrict__ W3,
                              _Float16* __restrict__ bp) {
    int tid = blockIdx.x * blockDim.x + threadIdx.x;
    if (tid >= NKT * NT_N * 32 * 16) return;
    int e    = tid & 15;
    int lane = (tid >> 4) & 31;
    int rest = tid >> 9;          // kt*NT_N + nt
    int nt   = rest % NT_N;
    int kt   = rest / NT_N;
    int kb = ((lane < 16) ? 0 : 8) + ((e < 8) ? 0 : 16);
    int k  = kt * 32 + kb + (e & 7);
    int n  = nt * 16 + (lane & 15);
    bp[tid] = (_Float16)wbig_val(k, n, W1, W2, W3);
}

__global__ void pack_bias_kernel(const float* __restrict__ b1,
                                 const float* __restrict__ b2,
                                 const float* __restrict__ b3,
                                 float* __restrict__ bias) {
    int i = blockIdx.x * blockDim.x + threadIdx.x;
    if (i >= 304) return;
    float v = 0.0f;
    if (i < 100)      v = b1[i];
    else if (i < 200) v = b2[i - 100];
    else if (i < 300) v = b3[i - 200];
    bias[i] = v;
}

// ---------------- WMMA GEMM: out = [T0|T1|T2|T3] @ Wbig + bias ----------------

__global__ void __launch_bounds__(256)
cheb_gemm_wmma(const float* __restrict__ T0, const float* __restrict__ T1,
               const float* __restrict__ T2, const float* __restrict__ T3,
               const _Float16* __restrict__ Bp, const float* __restrict__ bias,
               float* __restrict__ out) {
    const int lane = threadIdx.x & 31;
    const int wave = threadIdx.x >> 5;
    const int w    = blockIdx.x * 8 + wave;
    if (w >= TOTAL_TILES) return;
    const int nt = w % NT_N;
    const int mt = w / NT_N;
    const int m   = lane & 15;
    const int row = mt * 16 + m;                 // A-matrix row for this lane
    const int kb1 = (lane < 16) ? 0 : 8;         // ISA A-layout K base
    const float* Ts[4] = {T0, T1, T2, T3};

    v8f acc = {};
#pragma unroll
    for (int kt = 0; kt < NKT; ++kt) {
        const float* S = Ts[kt >> 1];            // which T matrix
        const int kl   = (kt & 1) * 32;          // K offset within that T
        const float* ap = S + (size_t)row * IC + kl + kb1;
        float4 a0 = *reinterpret_cast<const float4*>(ap);       // K = kb1+0..3
        float4 a1 = *reinterpret_cast<const float4*>(ap + 4);   // K = kb1+4..7
        float4 a2 = *reinterpret_cast<const float4*>(ap + 16);  // K = kb1+16..19
        float4 a3 = *reinterpret_cast<const float4*>(ap + 20);  // K = kb1+20..23
        v16h a;
        a[0]  = (_Float16)a0.x; a[1]  = (_Float16)a0.y;
        a[2]  = (_Float16)a0.z; a[3]  = (_Float16)a0.w;
        a[4]  = (_Float16)a1.x; a[5]  = (_Float16)a1.y;
        a[6]  = (_Float16)a1.z; a[7]  = (_Float16)a1.w;
        a[8]  = (_Float16)a2.x; a[9]  = (_Float16)a2.y;
        a[10] = (_Float16)a2.z; a[11] = (_Float16)a2.w;
        a[12] = (_Float16)a3.x; a[13] = (_Float16)a3.y;
        a[14] = (_Float16)a3.z; a[15] = (_Float16)a3.w;
        v16h b = *reinterpret_cast<const v16h*>(
            Bp + ((size_t)(kt * NT_N + nt) * 32 + lane) * 16);
        acc = __builtin_amdgcn_wmma_f32_16x16x32_f16(
            false, a, false, b, (short)0, acc, false, false);
    }

    const int col = nt * 16 + m;
    if (col < NCOL) {
        const float bv   = bias[col];
        const int rbase  = mt * 16 + ((lane < 16) ? 0 : 8);
#pragma unroll
        for (int r = 0; r < 8; ++r)
            out[(size_t)(rbase + r) * NCOL + col] = acc[r] + bv;
    }
}

// ---------------- host launch ----------------

extern "C" void kernel_launch(void* const* d_in, const int* in_sizes, int n_in,
                              void* d_out, int out_size, void* d_ws, size_t ws_size,
                              hipStream_t stream) {
    (void)in_sizes; (void)n_in; (void)out_size; (void)ws_size;
    const float*     x  = (const float*)d_in[0];
    const long long* ei = (const long long*)d_in[1];
    const float*     ew = (const float*)d_in[2];
    const float*     W1 = (const float*)d_in[3];
    const float*     b1 = (const float*)d_in[4];
    const float*     W2 = (const float*)d_in[5];
    const float*     b2 = (const float*)d_in[6];
    const float*     W3 = (const float*)d_in[7];
    const float*     b3 = (const float*)d_in[8];
    float* out = (float*)d_out;

    char* ws = (char*)d_ws;
    size_t off = 0;
    auto alloc = [&](size_t bytes) -> void* {
        void* p = ws + off;
        off = (off + bytes + 255) & ~(size_t)255;
        return p;
    };
    float*     deg  = (float*)alloc((size_t)NN * 4);
    float*     nrm  = (float*)alloc((size_t)NE * 4);
    float*     T1   = (float*)alloc((size_t)NN * IC * 4);
    float*     T2   = (float*)alloc((size_t)NN * IC * 4);
    float*     T3   = (float*)alloc((size_t)NN * IC * 4);
    _Float16*  Bp   = (_Float16*)alloc((size_t)NKT * NT_N * 32 * 16 * 2);
    float*     bias = (float*)alloc(304 * 4);

    const int eb  = (NE + 255) / 256;
    const int sb  = (int)(((long long)NE * 16 + 255) / 256);
    const int nb4 = (NN * IC / 4 + 255) / 256;

    hipMemsetAsync(deg, 0, (size_t)NN * 4, stream);
    hipMemsetAsync(T1,  0, (size_t)NN * IC * 4, stream);

    deg_kernel <<<eb, 256, 0, stream>>>(ei, ew, deg);
    norm_kernel<<<eb, 256, 0, stream>>>(ei, ew, deg, nrm);

    // T1 = P(x)
    scatter_kernel<<<sb, 256, 0, stream>>>(ei, nrm, x, T1, 1.0f);
    // T2 = 2*P(T1) - T0
    negcopy_kernel<<<nb4, 256, 0, stream>>>(x, T2, NN * IC / 4);
    scatter_kernel<<<sb, 256, 0, stream>>>(ei, nrm, T1, T2, 2.0f);
    // T3 = 2*P(T2) - T1
    negcopy_kernel<<<nb4, 256, 0, stream>>>(T1, T3, NN * IC / 4);
    scatter_kernel<<<sb, 256, 0, stream>>>(ei, nrm, T2, T3, 2.0f);

    pack_b_kernel<<<(NKT * NT_N * 32 * 16 + 255) / 256, 256, 0, stream>>>(W1, W2, W3, Bp);
    pack_bias_kernel<<<2, 256, 0, stream>>>(b1, b2, b3, bias);

    cheb_gemm_wmma<<<(TOTAL_TILES + 7) / 8, 256, 0, stream>>>(x, T1, T2, T3, Bp, bias, out);
}
